// RigidDenoiser_8169027797688
// MI455X (gfx1250) — matrix operand order repaired and checked by reading, not compiled, not run on hardware
//
#include <hip/hip_runtime.h>
#include <hip/hip_bf16.h>

typedef _Float16 f16;
typedef __attribute__((ext_vector_type(16))) _Float16 v16h;
typedef __attribute__((ext_vector_type(8)))  _Float16 v8h;
typedef __attribute__((ext_vector_type(8)))  float    v8f;

#define NRES 256
#define M_TOT (NRES*NRES)   // 65536 pair positions
#define LDA 40              // LDS row stride (halves): 80B, 16B-aligned, conflict-free
#define LDB 40

enum { SM_NORMAL=0, SM_CHMAJOR=1, SM_CHMAJOR_T=2, SM_QKV=3, SM_QKV_T=4,
       SM_ADD=5, SM_ADD_T=6, SM_NORMAL_H=7 };
enum { ACT_NONE=0, ACT_SIG=1, ACT_RELU=2 };

__device__ __forceinline__ float sigf(float x){ return 1.f/(1.f+__expf(-x)); }

__device__ __forceinline__ v8f wmma16(v16h a, v16h b, v8f c){
  return __builtin_amdgcn_wmma_f32_16x16x32_f16(false, a, false, b, (short)0, c, false, false);
}

__device__ __forceinline__ v16h join16(v8h lo, v8h hi){
  v16h r;
  #pragma unroll
  for (int e=0;e<8;++e){ r[e]=lo[e]; r[8+e]=hi[e]; }
  return r;
}

// ---------------------------------------------------------------------------
// Generic WMMA GEMM: out = act(A[M,K] @ W[K,N] + bias) (*emul) (*rowscale)
// A is f32 (optionally elementwise-multiplied by Amul) or f16 (Ah).
// Block: 256 threads (8 waves), tile 128x64, K-step 32.
// LDS: As row-major [128][LDA], Bs fragment-major [col][k] -> all fragment
// reads are aligned ds_load_b128. For K<=128 the whole weight tile is staged
// once (4 chunks resident); K=512 falls back to per-chunk staging.
// ---------------------------------------------------------------------------
__global__ __launch_bounds__(256) void gemm_kernel(
    const float* __restrict__ A, const f16* __restrict__ Ah, const float* __restrict__ Amul,
    const float* __restrict__ W, const float* __restrict__ bias,
    float* __restrict__ outF, f16* __restrict__ outH,
    const float* __restrict__ emul, const float* __restrict__ rowscale,
    int M, int N, int K, int act, int smode)
{
  __shared__ __align__(32) f16 As[128*LDA];
  __shared__ __align__(32) f16 Bs[4*64*LDB];
  const int tid  = threadIdx.x;
  const int wv   = tid >> 5;
  const int lane = tid & 31;
  const int gm0  = blockIdx.y * 128;
  const int gn0  = blockIdx.x * 64;
  const int m    = lane & 15, g = lane >> 4;
  const char* pfb = Ah ? (const char*)Ah : (const char*)A;   // prefetch base
  const int   pfe = Ah ? 1 : 2;                               // log2(elem size)
  v8f acc[4] = {};

  // staging assignments (one chunk per thread)
  const int ar = tid >> 1, ac0 = (tid & 1) * 16;   // A: 16 halves each
  const int br = tid & 31, bc0 = (tid >> 5) * 8;   // B: 8 cols each

  const int  kch  = K >> 5;
  const bool ball = (kch <= 4);     // whole weight tile resident in LDS

  // ---- stage B chunk `src` (32 x 64) fragment-major into chunk slot `dst` ----
  auto stageB = [&](int dst, int src) {
    f16* bs = Bs + dst*64*LDB;
    const int kk0 = src*32;
    if (gn0 + 64 <= N) {
      const float4* pw = (const float4*)(W + (size_t)(kk0 + br)*N + gn0 + bc0);
      float4 w0 = pw[0], w1 = pw[1];
      bs[(bc0+0)*LDB + br] = (f16)w0.x;
      bs[(bc0+1)*LDB + br] = (f16)w0.y;
      bs[(bc0+2)*LDB + br] = (f16)w0.z;
      bs[(bc0+3)*LDB + br] = (f16)w0.w;
      bs[(bc0+4)*LDB + br] = (f16)w1.x;
      bs[(bc0+5)*LDB + br] = (f16)w1.y;
      bs[(bc0+6)*LDB + br] = (f16)w1.z;
      bs[(bc0+7)*LDB + br] = (f16)w1.w;
    } else {
      #pragma unroll
      for (int e=0;e<8;++e){
        int gc = gn0 + bc0 + e;
        bs[(bc0+e)*LDB + br] = (f16)((gc < N) ? W[(size_t)(kk0 + br)*N + gc] : 0.f);
      }
    }
  };

  if (ball) {
    for (int ck = 0; ck < kch; ++ck) stageB(ck, ck);
  }

  for (int ck = 0; ck < kch; ++ck) {
    const int k0 = ck*32;
    if (ck + 1 < kch)   // global_prefetch_b8 of next A K-tile
      __builtin_prefetch(pfb + (((size_t)(gm0 + ar) * K + k0 + 32) << pfe));

    // ---- stage A (128 x 32) as f16 ----
    {
      f16* dst = As + ar*LDA + ac0;
      if (Ah) {
        const v8h* pa = (const v8h*)(Ah + (size_t)(gm0 + ar)*K + k0 + ac0);
        ((v8h*)dst)[0] = pa[0];
        ((v8h*)dst)[1] = pa[1];
      } else {
        const float4* pa = (const float4*)(A + (size_t)(gm0 + ar)*K + k0 + ac0);
        float4 f0 = pa[0], f1 = pa[1], f2 = pa[2], f3 = pa[3];
        if (Amul) {
          const float4* pm4 = (const float4*)(Amul + (size_t)(gm0 + ar)*K + k0 + ac0);
          float4 m0 = pm4[0], m1 = pm4[1], m2 = pm4[2], m3 = pm4[3];
          f0.x*=m0.x; f0.y*=m0.y; f0.z*=m0.z; f0.w*=m0.w;
          f1.x*=m1.x; f1.y*=m1.y; f1.z*=m1.z; f1.w*=m1.w;
          f2.x*=m2.x; f2.y*=m2.y; f2.z*=m2.z; f2.w*=m2.w;
          f3.x*=m3.x; f3.y*=m3.y; f3.z*=m3.z; f3.w*=m3.w;
        }
        v8h x0, x1;
        x0[0]=(f16)f0.x; x0[1]=(f16)f0.y; x0[2]=(f16)f0.z; x0[3]=(f16)f0.w;
        x0[4]=(f16)f1.x; x0[5]=(f16)f1.y; x0[6]=(f16)f1.z; x0[7]=(f16)f1.w;
        x1[0]=(f16)f2.x; x1[1]=(f16)f2.y; x1[2]=(f16)f2.z; x1[3]=(f16)f2.w;
        x1[4]=(f16)f3.x; x1[5]=(f16)f3.y; x1[6]=(f16)f3.z; x1[7]=(f16)f3.w;
        ((v8h*)dst)[0] = x0;
        ((v8h*)dst)[1] = x1;
      }
    }
    if (!ball) stageB(0, ck);
    __syncthreads();

    // ---- fragments + WMMA ----
    const f16* pa = As + (wv*16 + m)*LDA + g*8;
    v16h af = join16(*(const v8h*)pa, *(const v8h*)(pa + 16));
    const f16* bsb = Bs + (ball ? ck : 0)*64*LDB;
    #pragma unroll
    for (int nt=0; nt<4; ++nt) {
      const f16* pb = bsb + (nt*16 + m)*LDB + g*8;
      v16h bf = join16(*(const v8h*)pb, *(const v8h*)(pb + 16));
      acc[nt] = wmma16(af, bf, acc[nt]);
    }
    __syncthreads();
  }

  // ---- epilogue ----
  #pragma unroll
  for (int nt=0; nt<4; ++nt) {
    int col = gn0 + nt*16 + m;
    if (col >= N) continue;
    #pragma unroll
    for (int r=0;r<8;++r){
      int row = gm0 + wv*16 + r + g*8;
      float v = acc[nt][r];
      if (bias) v += bias[col];
      if (act == ACT_SIG) v = sigf(v);
      else if (act == ACT_RELU) v = fmaxf(v, 0.f);
      if (emul) v *= emul[(size_t)row*N + col];
      switch (smode) {
        case SM_NORMAL:   outF[(size_t)row*N + col] = v; break;
        case SM_NORMAL_H: outH[(size_t)row*N + col] = (f16)v; break;
        case SM_CHMAJOR: {
          float sc = rowscale ? rowscale[row] : 1.f;
          outH[(size_t)col*M + row] = (f16)(v*sc);
        } break;
        case SM_CHMAJOR_T: {
          float sc = rowscale ? rowscale[row] : 1.f;
          outH[(size_t)col*M + (size_t)(row & 255)*256 + (row >> 8)] = (f16)(v*sc);
        } break;
        case SM_QKV: {
          int i = row >> 8, j = row & 255, h = col >> 5, ca = col & 31;
          outH[(((size_t)(i*4 + h))*256 + j)*32 + ca] = (f16)v;
        } break;
        case SM_QKV_T: {
          int i = row >> 8, j = row & 255, h = col >> 5, ca = col & 31;
          outH[(((size_t)(i*4 + h))*32 + ca)*256 + j] = (f16)v;
        } break;
        case SM_ADD: {
          float sc = rowscale ? rowscale[row] : 1.f;
          outF[(size_t)row*128 + col] += v*sc;
        } break;
        case SM_ADD_T: {
          int q = (row & 255)*256 + (row >> 8);
          float sc = rowscale ? rowscale[q] : 1.f;
          outF[(size_t)q*128 + col] += v*sc;
        } break;
      }
    }
  }
}

// ---------------------------------------------------------------------------
// Triangle einsum: out[i,j,c] = sum_k a[c][i][k] * b[c][j][k]
// a,b packed channel-major f16. One wave per (16x16 tile, channel); 8 WMMAs.
// ---------------------------------------------------------------------------
__global__ __launch_bounds__(32) void tri_kernel(const f16* __restrict__ ap,
    const f16* __restrict__ bp, float* __restrict__ out)
{
  const int it = blockIdx.x, jt = blockIdx.y, c = blockIdx.z;
  const int lane = threadIdx.x;
  const int m = lane & 15, g = lane >> 4;
  const f16* ac = ap + (size_t)c * M_TOT;
  const f16* bc = bp + (size_t)c * M_TOT;
  v8f acc = {};
  #pragma unroll
  for (int k0 = 0; k0 < 256; k0 += 32) {
    const f16* pa = ac + (size_t)(it*16 + m)*256 + k0 + g*8;
    v16h af = join16(*(const v8h*)pa, *(const v8h*)(pa + 16));
    const f16* pb = bc + (size_t)(jt*16 + m)*256 + k0 + g*8;
    v16h bf = join16(*(const v8h*)pb, *(const v8h*)(pb + 16));
    acc = wmma16(af, bf, acc);
  }
  const int col = jt*16 + m;
  #pragma unroll
  for (int r=0;r<8;++r){
    int rowi = it*16 + r + g*8;
    out[((size_t)rowi*256 + col)*128 + c] = acc[r];
  }
}

// ---------------------------------------------------------------------------
// Flash triangle attention for one (i, head): softmax_k(q.kT/sqrt(32)+bias+mask).v
// q,k packed [i][h][j][32] f16, v packed [i][h][32][j] f16, o -> [p][h*32+ca] f32.
// 8 waves; each wave owns 2 J-tiles; streaming softmax over K in steps of 32.
// ---------------------------------------------------------------------------
__global__ __launch_bounds__(256) void flash_kernel(
    const f16* __restrict__ qp, const f16* __restrict__ kp, const f16* __restrict__ vp,
    const float* __restrict__ xb, const float* __restrict__ mask, float* __restrict__ o)
{
  const int i = blockIdx.x, h = blockIdx.y;
  const int tid = threadIdx.x, wave = tid >> 5, lane = tid & 31;
  const int n = lane & 15, g = lane >> 4;
  const f16* qb = qp + ((size_t)(i*4 + h))*256*32;
  const f16* kb = kp + ((size_t)(i*4 + h))*256*32;
  const f16* vb = vp + ((size_t)(i*4 + h))*32*256;
  __shared__ __align__(32) f16 Ps[8*16*LDA];
  f16* myP = Ps + wave*16*LDA;
  const float rs = 0.17677669529663687f;   // 1/sqrt(32)

  for (int jj = 0; jj < 2; ++jj) {
    const int jt = wave*2 + jj;
    const f16* pq = qb + (size_t)(jt*16 + n)*32 + g*8;
    v16h qf = join16(*(const v8h*)pq, *(const v8h*)(pq + 16));
    v8f o0 = {}, o1 = {};
    float Mr[8], Lr[8];
    #pragma unroll
    for (int r=0;r<8;++r){ Mr[r] = -1e30f; Lr[r] = 0.f; }

    for (int k2 = 0; k2 < 8; ++k2) {
      const int kbase = k2*32;
      const f16* pk0 = kb + (size_t)(kbase      + n)*32 + g*8;
      const f16* pk1 = kb + (size_t)(kbase + 16 + n)*32 + g*8;
      v16h kf0 = join16(*(const v8h*)pk0, *(const v8h*)(pk0 + 16));
      v16h kf1 = join16(*(const v8h*)pk1, *(const v8h*)(pk1 + 16));
      v8f s0 = {}, s1 = {};
      s0 = wmma16(qf, kf0, s0);
      s1 = wmma16(qf, kf1, s1);
      const int kc0 = kbase + n, kc1 = kc0 + 16;
      const float mt0 = 1e9f*(mask[i*256 + kc0] - 1.f);
      const float mt1 = 1e9f*(mask[i*256 + kc1] - 1.f);
      #pragma unroll
      for (int r=0;r<8;++r){
        const int j = jt*16 + r + g*8;
        float v0 = s0[r]*rs + mt0 + xb[((size_t)j*256 + kc0)*4 + h];
        float v1 = s1[r]*rs + mt1 + xb[((size_t)j*256 + kc1)*4 + h];
        float mx = fmaxf(v0, v1);
        #pragma unroll
        for (int d=1; d<16; d<<=1) mx = fmaxf(mx, __shfl_xor(mx, d, 32));
        const float nm = fmaxf(Mr[r], mx);
        const float sc = __expf(Mr[r] - nm);
        const float p0 = __expf(v0 - nm), p1 = __expf(v1 - nm);
        float rsum = p0 + p1;
        #pragma unroll
        for (int d=1; d<16; d<<=1) rsum += __shfl_xor(rsum, d, 32);
        Lr[r] = Lr[r]*sc + rsum; Mr[r] = nm;
        o0[r] *= sc; o1[r] *= sc;
        const int mrow = r + g*8;
        myP[mrow*LDA + n]      = (f16)p0;
        myP[mrow*LDA + 16 + n] = (f16)p1;
      }
      __syncthreads();
      const f16* pp = myP + n*LDA + g*8;
      v16h pf = join16(*(const v8h*)pp, *(const v8h*)(pp + 16));
      const f16* pv0 = vb + (size_t)n*256        + kbase + g*8;
      const f16* pv1 = vb + (size_t)(16 + n)*256 + kbase + g*8;
      v16h vf0 = join16(*(const v8h*)pv0, *(const v8h*)(pv0 + 16));
      v16h vf1 = join16(*(const v8h*)pv1, *(const v8h*)(pv1 + 16));
      o0 = wmma16(pf, vf0, o0);
      o1 = wmma16(pf, vf1, o1);
      __syncthreads();
    }
    #pragma unroll
    for (int r=0;r<8;++r){
      const float inv = 1.f/fmaxf(Lr[r], 1e-20f);
      const int j = jt*16 + r + g*8;
      size_t base = ((size_t)i*256 + j)*128 + h*32;
      o[base + n]      = o0[r]*inv;
      o[base + 16 + n] = o1[r]*inv;
    }
  }
}

// ---------------------------------------------------------------------------
// Elementwise kernels (wave-per-row LayerNorm over C=128)
// ---------------------------------------------------------------------------
__global__ __launch_bounds__(256) void ln_kernel(const float* __restrict__ x,
    const float* __restrict__ gw, const float* __restrict__ bw,
    float* __restrict__ y, int mode)   // 0: none, 1: *g, 2: *g+b
{
  const int row = blockIdx.x*8 + (threadIdx.x >> 5);
  const int lane = threadIdx.x & 31;
  const float4 v = ((const float4*)(x + (size_t)row*128))[lane];
  float s = v.x+v.y+v.z+v.w;
  #pragma unroll
  for (int m=1;m<32;m<<=1) s += __shfl_xor(s, m, 32);
  const float mu = s * (1.f/128.f);
  const float d0=v.x-mu, d1=v.y-mu, d2=v.z-mu, d3=v.w-mu;
  float q = d0*d0+d1*d1+d2*d2+d3*d3;
  #pragma unroll
  for (int m=1;m<32;m<<=1) q += __shfl_xor(q, m, 32);
  const float rinv = rsqrtf(q*(1.f/128.f) + 1e-5f);
  const int c = lane*4;
  float4 o; o.x=d0*rinv; o.y=d1*rinv; o.z=d2*rinv; o.w=d3*rinv;
  if (mode >= 1) { o.x*=gw[c]; o.y*=gw[c+1]; o.z*=gw[c+2]; o.w*=gw[c+3]; }
  if (mode == 2) { o.x+=bw[c]; o.y+=bw[c+1]; o.z+=bw[c+2]; o.w+=bw[c+3]; }
  ((float4*)(y + (size_t)row*128))[lane] = o;
}

// dst = gate * LN_noaffine(z) + dst      (AdaLN combine; dst holds shift)
__global__ __launch_bounds__(256) void adaln_combine_kernel(const float* __restrict__ z,
    const float* __restrict__ gate, float* __restrict__ dst)
{
  const int row = blockIdx.x*8 + (threadIdx.x >> 5);
  const int lane = threadIdx.x & 31;
  const float4 v = ((const float4*)(z + (size_t)row*128))[lane];
  float s = v.x+v.y+v.z+v.w;
  #pragma unroll
  for (int m=1;m<32;m<<=1) s += __shfl_xor(s, m, 32);
  const float mu = s * (1.f/128.f);
  const float d0=v.x-mu, d1=v.y-mu, d2=v.z-mu, d3=v.w-mu;
  float q = d0*d0+d1*d1+d2*d2+d3*d3;
  #pragma unroll
  for (int m=1;m<32;m<<=1) q += __shfl_xor(q, m, 32);
  const float rinv = rsqrtf(q*(1.f/128.f) + 1e-5f);
  const float4 gt = ((const float4*)(gate + (size_t)row*128))[lane];
  float4 d = ((float4*)(dst + (size_t)row*128))[lane];
  d.x = gt.x*(d0*rinv) + d.x;
  d.y = gt.y*(d1*rinv) + d.y;
  d.z = gt.z*(d2*rinv) + d.z;
  d.w = gt.w*(d3*rinv) + d.w;
  ((float4*)(dst + (size_t)row*128))[lane] = d;
}

__global__ void copy_kernel(const float4* __restrict__ src, float4* __restrict__ dst, int n4){
  int i = blockIdx.x*256 + threadIdx.x;
  if (i < n4) dst[i] = src[i];
}

__global__ void transpose_c_kernel(const float* __restrict__ src, float* __restrict__ dst){
  int idx = blockIdx.x*256 + threadIdx.x;   // M_TOT*128 threads
  int c = idx & 127, p = idx >> 7, i = p >> 8, j = p & 255;
  dst[(((size_t)j*256 + i) << 7) + c] = src[idx];
}

__global__ void transpose_m_kernel(const float* __restrict__ src, float* __restrict__ dst){
  int idx = blockIdx.x*256 + threadIdx.x;
  if (idx < M_TOT) dst[(idx & 255)*256 + (idx >> 8)] = src[idx];
}

// ---------------------------------------------------------------------------
// Host side
// ---------------------------------------------------------------------------
static inline void launch_gemm(hipStream_t st, const float* A, const f16* Ah, const float* Amul,
    const float* W, const float* bias, float* outF, f16* outH,
    const float* emul, const float* rowscale, int N, int K, int act, int smode)
{
  dim3 grid((N + 63)/64, M_TOT/128);
  gemm_kernel<<<grid, 256, 0, st>>>(A, Ah, Amul, W, bias, outF, outH,
                                    emul, rowscale, M_TOT, N, K, act, smode);
}

struct AttP { const float *cln_g,*gate_b,*gate_w,*shift_w,*b_w,*g_b,*g_w,*k_w,*o_b,*o_w,*q_w,*v_w; };
struct MulP { const float *ag_b,*ag_w,*ap_b,*ap_w,*bg_b,*bg_w,*bp_b,*bp_w,*g_b,*g_w,
              *ln_in_b,*ln_in_g,*ln_out_b,*ln_out_g,*zp_b,*zp_w; };
struct TrsP { const float *cln_g,*gate_b,*gate_w,*shift_w,*b1,*b2,*ln_b,*ln_g,
              *out_gate_b,*out_gate_w,*w1,*w2; };
struct WSP { float *Z0,*Z1,*Z2,*Z3,*Z4,*Z5,*XB,*MT; f16 *H0,*H1,*H2; };

static void run_tri_mul(hipStream_t st, const MulP& P, float* s,
                        const float* mask, const float* icm, const WSP& w, bool outgoing)
{
  ln_kernel<<<8192,256,0,st>>>(s, P.ln_in_g, P.ln_in_b, w.Z0, 2);
  const int apk = outgoing ? SM_CHMAJOR : SM_CHMAJOR_T;
  launch_gemm(st, w.Z0,0,0, P.ag_w, P.ag_b, w.Z1,0, 0,0, 128,128, ACT_SIG, SM_NORMAL);
  launch_gemm(st, w.Z0,0,0, P.ap_w, P.ap_b, 0,w.H0, w.Z1, mask, 128,128, ACT_NONE, apk);
  launch_gemm(st, w.Z0,0,0, P.bg_w, P.bg_b, w.Z1,0, 0,0, 128,128, ACT_SIG, SM_NORMAL);
  launch_gemm(st, w.Z0,0,0, P.bp_w, P.bp_b, 0,w.H1, w.Z1, mask, 128,128, ACT_NONE, apk);
  tri_kernel<<<dim3(16,16,128),32,0,st>>>(w.H0, w.H1, w.Z2);
  ln_kernel<<<8192,256,0,st>>>(w.Z2, P.ln_out_g, P.ln_out_b, w.Z1, 2);
  launch_gemm(st, w.Z0,0,0, P.g_w, P.g_b, w.Z3,0, 0,0, 128,128, ACT_SIG, SM_NORMAL);
  launch_gemm(st, w.Z1,0,0, P.zp_w, P.zp_b, s,0, w.Z3, icm, 128,128, ACT_NONE, SM_ADD);
}

static void run_tri_att(hipStream_t st, const AttP& P, const float* zsrc, const float* cond,
                        const float* mask, const float* icm, float* s, const WSP& w, bool ending)
{
  ln_kernel<<<8192,256,0,st>>>(cond, P.cln_g, 0, w.Z1, 1);
  launch_gemm(st, w.Z1,0,0, P.gate_w, P.gate_b, w.Z3,0, 0,0, 128,128, ACT_SIG, SM_NORMAL);
  launch_gemm(st, w.Z1,0,0, P.shift_w, 0, w.Z0,0, 0,0, 128,128, ACT_NONE, SM_NORMAL);
  adaln_combine_kernel<<<8192,256,0,st>>>(zsrc, w.Z3, w.Z0);
  launch_gemm(st, w.Z0,0,0, P.q_w, 0, 0,w.H0, 0,0, 128,128, ACT_NONE, SM_QKV);
  launch_gemm(st, w.Z0,0,0, P.k_w, 0, 0,w.H1, 0,0, 128,128, ACT_NONE, SM_QKV);
  launch_gemm(st, w.Z0,0,0, P.v_w, 0, 0,w.H2, 0,0, 128,128, ACT_NONE, SM_QKV_T);
  launch_gemm(st, w.Z0,0,0, P.b_w, 0, w.XB,0, 0,0, 4,128, ACT_NONE, SM_NORMAL);
  launch_gemm(st, w.Z0,0,0, P.g_w, P.g_b, w.Z3,0, 0,0, 128,128, ACT_SIG, SM_NORMAL);
  flash_kernel<<<dim3(256,4),256,0,st>>>(w.H0, w.H1, w.H2, w.XB, mask, w.Z2);
  launch_gemm(st, w.Z2,0, w.Z3, P.o_w, P.o_b, s,0, 0, icm, 128,128, ACT_NONE,
              ending ? SM_ADD_T : SM_ADD);
}

extern "C" void kernel_launch(void* const* d_in, const int* in_sizes, int n_in,
                              void* d_out, int out_size, void* d_ws, size_t ws_size,
                              hipStream_t stream)
{
  (void)in_sizes; (void)n_in; (void)out_size; (void)ws_size;
  const float* z    = (const float*)d_in[0];
  const float* cond = (const float*)d_in[1];
  const float* pm   = (const float*)d_in[2];
  const float* icm  = (const float*)d_in[3];
  auto F = [&](int i){ return (const float*)d_in[i]; };

  // params pytree: top-level dict (att_end, att_start, mul_in, mul_out, trans),
  // nested dicts sorted alphabetically -> flat leaf order:
  AttP att_end  { F(4),F(5),F(6),F(7),F(8),F(9),F(10),F(11),F(12),F(13),F(14),F(15) };
  AttP att_start{ F(16),F(17),F(18),F(19),F(20),F(21),F(22),F(23),F(24),F(25),F(26),F(27) };
  MulP mul_in   { F(28),F(29),F(30),F(31),F(32),F(33),F(34),F(35),F(36),F(37),
                  F(38),F(39),F(40),F(41),F(42),F(43) };
  MulP mul_out  { F(44),F(45),F(46),F(47),F(48),F(49),F(50),F(51),F(52),F(53),
                  F(54),F(55),F(56),F(57),F(58),F(59) };
  TrsP trans    { F(60),F(61),F(62),F(63),F(64),F(65),F(66),F(67),F(68),F(69),F(70),F(71) };

  char* wsb = (char*)d_ws;
  const size_t SZF = (size_t)M_TOT * 128 * 4;
  WSP w;
  w.Z0 = (float*)(wsb);
  w.Z1 = (float*)(wsb + SZF);
  w.Z2 = (float*)(wsb + 2*SZF);
  w.Z3 = (float*)(wsb + 3*SZF);
  w.Z4 = (float*)(wsb + 4*SZF);
  w.Z5 = (float*)(wsb + 5*SZF);
  w.XB = (float*)(wsb + 6*SZF);
  w.MT = (float*)(wsb + 6*SZF + (size_t)M_TOT*4*4);
  w.H0 = (f16*)(wsb + 6*SZF + (size_t)2*1024*1024);
  w.H1 = w.H0 + (size_t)M_TOT*128;
  w.H2 = w.H1 + (size_t)M_TOT*128;

  float* s = (float*)d_out;

  // s = z
  copy_kernel<<<8192,256,0,stream>>>((const float4*)z, (float4*)s, M_TOT*32);

  // triangle multiplicative: outgoing, then incoming
  run_tri_mul(stream, mul_out, s, pm, icm, w, true);
  run_tri_mul(stream, mul_in,  s, pm, icm, w, false);

  // triangle attention, starting node
  run_tri_att(stream, att_start, s, cond, pm, icm, s, w, false);

  // triangle attention, ending node: transpose s / cond / mask
  transpose_c_kernel<<<32768,256,0,stream>>>(s, w.Z4);
  transpose_c_kernel<<<32768,256,0,stream>>>(cond, w.Z5);
  transpose_m_kernel<<<256,256,0,stream>>>(pm, w.MT);
  run_tri_att(stream, att_end, w.Z4, w.Z5, w.MT, icm, s, w, true);

  // conditioned (AdaLN-zero) pair transition
  ln_kernel<<<8192,256,0,stream>>>(cond, trans.cln_g, 0, w.Z1, 1);
  launch_gemm(stream, w.Z1,0,0, trans.gate_w, trans.gate_b, w.Z3,0, 0,0, 128,128, ACT_SIG, SM_NORMAL);
  launch_gemm(stream, w.Z1,0,0, trans.shift_w, 0, w.Z0,0, 0,0, 128,128, ACT_NONE, SM_NORMAL);
  adaln_combine_kernel<<<8192,256,0,stream>>>(s, w.Z3, w.Z0);
  ln_kernel<<<8192,256,0,stream>>>(w.Z0, trans.ln_g, trans.ln_b, w.Z1, 2);
  launch_gemm(stream, w.Z1,0,0, trans.w1, trans.b1, 0,w.H0, 0,0, 512,128, ACT_RELU, SM_NORMAL_H);
  launch_gemm(stream, cond,0,0, trans.out_gate_w, trans.out_gate_b, w.Z3,0, 0,0, 128,128, ACT_SIG, SM_NORMAL);
  launch_gemm(stream, 0,w.H0,0, trans.w2, trans.b2, s,0, w.Z3, icm, 128,512, ACT_NONE, SM_ADD);
}
